// VectorQuantizerEMA_43044162240852
// MI455X (gfx1250) — matrix operand order repaired and checked by reading, not compiled, not run on hardware
//
#include <hip/hip_runtime.h>
#include <hip/hip_bf16.h>
#include <stdint.h>

// ---------------- problem constants ----------------
#define DECAY_F 0.99f
#define EPS_F   1e-7f
#define CCOST_F 0.25f

constexpr int Dn = 1024, Kn = 8192, Nn = 8 * 2048;     // D, K, N=B*S
constexpr long long DK = (long long)Dn * Kn;           // 8,388,608

// output layout (flat fp32 elements; indices are int32 bit-stored)
constexpr long long OUT_LAT  = 0;
constexpr long long OUT_LOSS = (long long)Nn * Dn;     // 16,777,216
constexpr long long OUT_IDX  = OUT_LOSS + 1;
constexpr long long OUT_VQ   = OUT_IDX + Nn;
constexpr long long OUT_EMA  = OUT_VQ + DK;
constexpr long long OUT_CS   = OUT_EMA + DK;

// workspace layout (bytes): dw | counts | nsum(+pad) | enorm | bfE[K][D] | bfQ[N][D]
constexpr size_t WSB_DW  = 0;
constexpr size_t WSB_CNT = WSB_DW + (size_t)DK * 4;
constexpr size_t WSB_NS  = WSB_CNT + (size_t)Kn * 4;
constexpr size_t WSB_EN  = WSB_NS + 16;                 // keep 16B alignment
constexpr size_t WSB_BFE = WSB_EN + (size_t)Kn * 4;     // bf16 codebook, [K][D]
constexpr size_t WSB_BFQ = WSB_BFE + (size_t)DK * 2;    // bf16 query, [N][D]

// ---------------- WMMA types ----------------
typedef __attribute__((ext_vector_type(16))) __bf16 v16bf;
typedef __attribute__((ext_vector_type(8)))  float  v8f;
typedef __attribute__((ext_vector_type(4)))  int    i4v;

union Frag { uint4 x[2]; unsigned u[8]; v16bf v; };

__device__ __forceinline__ uint16_t f2bf(float f) {
  unsigned u = __float_as_uint(f);
  unsigned r = u + 0x7FFFu + ((u >> 16) & 1u);   // round-to-nearest-even
  return (uint16_t)(r >> 16);
}

// gfx1250 async global->LDS path (guarded; falls back to plain copy)
#ifndef __has_builtin
#define __has_builtin(x) 0
#endif
#if __has_builtin(__builtin_amdgcn_global_load_async_to_lds_b128) && \
    __has_builtin(__builtin_amdgcn_s_wait_asynccnt)
#define HAVE_ASYNC 1
#else
#define HAVE_ASYNC 0
#endif
#define AS1 __attribute__((address_space(1)))
#define AS3 __attribute__((address_space(3)))

// ---------------- tiling ----------------
constexpr int MT   = 64;          // query rows per workgroup
constexpr int KT   = 64;          // codewords per K-tile
constexpr int QSTR = Dn + 8;      // padded LDS row stride (bf16 elems); 2064B = 16B-aligned

// ---------------- kernel 1: zero dw/counts/nsum + loss ----------------
__global__ void vq_zero(float* __restrict__ ws, float* __restrict__ out) {
  long long i = (long long)blockIdx.x * 256 + threadIdx.x;
  long long total = DK + Kn + 1;        // dw + counts + nsum (contiguous)
  if (i < total) ws[i] = 0.0f;
  if (i == 0) out[OUT_LOSS] = 0.0f;
}

// ---------------- kernel 2: codebook column norms ----------------
__global__ void vq_enorm(const float* __restrict__ vq, float* __restrict__ en) {
  int k = blockIdx.x * 256 + threadIdx.x;     // coalesced over k
  float s = 0.0f;
  for (int d = 0; d < Dn; ++d) {
    float v = vq[(long long)d * Kn + k];
    s = fmaf(v, v, s);
  }
  en[k] = s;
}

// ---------------- kernel 3a: query fp32 -> bf16 (same layout) -----------
__global__ void vq_cvtq(const float* __restrict__ q, uint16_t* __restrict__ bfQ) {
  long long j = ((long long)blockIdx.x * 256 + threadIdx.x) * 2;
  if (j >= (long long)Nn * Dn) return;
  float2 f = *(const float2*)(q + j);
  unsigned lo = f2bf(f.x), hi = (unsigned)f2bf(f.y);
  *(unsigned*)(bfQ + j) = lo | (hi << 16);
}

// ---------------- kernel 3b: codebook fp32 [D][K] -> bf16 [K][D] --------
__global__ __launch_bounds__(256) void vq_cvte(const float* __restrict__ vq,
                                               uint16_t* __restrict__ bfE) {
  __shared__ uint16_t tile[64 * 72];
  int k0 = (blockIdx.x & (Kn / 64 - 1)) * 64;
  int d0 = (blockIdx.x / (Kn / 64)) * 64;
  int tid = threadIdx.x;
  for (int i = 0; i < 16; ++i) {                 // read coalesced over k
    int idx = tid + i * 256;
    int c = idx & 63, r = idx >> 6;              // c -> k, r -> d
    tile[c * 72 + r] = f2bf(vq[(long long)(d0 + r) * Kn + (k0 + c)]);
  }
  __syncthreads();
  for (int i = 0; i < 8; ++i) {                  // write coalesced over d
    int idx = tid + i * 256;
    int dd = (idx & 31) * 2, kk = idx >> 5;
    unsigned lo = tile[kk * 72 + dd], hi = (unsigned)tile[kk * 72 + dd + 1];
    *(unsigned*)(bfE + (long long)(k0 + kk) * Dn + d0 + dd) = lo | (hi << 16);
  }
}

// ---------------- kernel 4: WMMA distances + argmin + gather + scatter --
__global__ __launch_bounds__(256) void vq_main(
    const float* __restrict__ query, const float* __restrict__ vq,
    const uint16_t* __restrict__ bfQ, const uint16_t* __restrict__ bfE,
    const float* __restrict__ en, float* __restrict__ out,
    float* __restrict__ dw, float* __restrict__ counts) {
  extern __shared__ char smem[];
  uint16_t* qlds  = (uint16_t*)smem;                               // 64*1032 bf16
  float*    redv  = (float*)(smem + (size_t)MT * QSTR * 2);
  int*      redi  = (int*)(redv + 2 * MT);
  int*      chosen= (int*)(redi + 2 * MT);
  float*    lred  = (float*)(chosen + MT);

  const int tid  = threadIdx.x;
  const int row0 = blockIdx.x * MT;
  const int w    = tid >> 5, lane = tid & 31;
  const int g    = lane >> 4, li = lane & 15;
  const int mw   = w >> 1, nw = w & 1;        // 4 M-waves x 2 N-waves

  // stage 64 bf16 query rows into LDS (async global->LDS where available)
  for (int i = 0; i < (MT * Dn) / (256 * 8); ++i) {   // 32 x b128 per thread
    int idx = tid + i * 256;
    int r = idx >> 7, q8 = (idx & 127) * 8;
    const uint16_t* src = bfQ + (long long)(row0 + r) * Dn + q8;
    uint16_t* dst = qlds + r * QSTR + q8;
#if HAVE_ASYNC
    __builtin_amdgcn_global_load_async_to_lds_b128((AS1 i4v*)src,
                                                   (AS3 i4v*)dst, 0, 0);
#else
    *(uint4*)dst = *(const uint4*)src;
#endif
  }
#if HAVE_ASYNC
  __builtin_amdgcn_s_wait_asynccnt(0);
#endif
  __syncthreads();

  float rv[8];
  int   ri[8];
#pragma unroll
  for (int v = 0; v < 8; ++v) { rv[v] = 3.4e38f; ri[v] = 0; }

  const int arow = mw * 16 + li;       // A-fragment row (lane = M)
  const int ncb0 = nw * 32;            // this wave's column base inside K-tile
  const uint16_t* aptr = qlds + arow * QSTR + (g << 3);

  for (int k0 = 0; k0 < Kn; k0 += KT) {
    v8f acc0 = {0,0,0,0,0,0,0,0};
    v8f acc1 = {0,0,0,0,0,0,0,0};
    // B fragments straight from L2-resident bf16 [K][D] codebook:
    // per lane two 16B runs at row (k0+ncb0[+16]+li), offsets g*8 and 16+g*8.
    const uint16_t* brow0 = bfE + (long long)(k0 + ncb0 + li) * Dn + (g << 3);
    const uint16_t* brow1 = brow0 + (long long)16 * Dn;
#pragma unroll 4
    for (int d0 = 0; d0 < Dn; d0 += 32) {
      Frag a, b0, b1;
      a.x[0]  = *(const uint4*)(aptr + d0);        // K = g*8 + 0..7
      a.x[1]  = *(const uint4*)(aptr + d0 + 16);   // K = 16 + g*8 + 0..7
      b0.x[0] = *(const uint4*)(brow0 + d0);       // K = g*8 + 0..7
      b0.x[1] = *(const uint4*)(brow0 + d0 + 16);  // K = 16 + g*8 + 0..7
      b1.x[0] = *(const uint4*)(brow1 + d0);
      b1.x[1] = *(const uint4*)(brow1 + d0 + 16);
      acc0 = __builtin_amdgcn_wmma_f32_16x16x32_bf16(false, a.v, false, b0.v,
                                                     (short)0, acc0, false, false);
      acc1 = __builtin_amdgcn_wmma_f32_16x16x32_bf16(false, a.v, false, b1.v,
                                                     (short)0, acc1, false, false);
    }
    // score = ||e||^2 - 2*dot ; running per-row argmin (strict < keeps first)
    float e0 = en[k0 + ncb0 + li];
    float e1 = en[k0 + ncb0 + 16 + li];
    int   c0 = k0 + ncb0 + li, c1 = c0 + 16;
#pragma unroll
    for (int v = 0; v < 8; ++v) {
      float s0 = fmaf(-2.0f, acc0[v], e0);
      float s1 = fmaf(-2.0f, acc1[v], e1);
      if (s0 < rv[v]) { rv[v] = s0; ri[v] = c0; }
      if (s1 < rv[v]) { rv[v] = s1; ri[v] = c1; }
    }
  }

  // argmin reduce across the 16 lanes of each half-wave (columns)
#pragma unroll
  for (int v = 0; v < 8; ++v) {
#pragma unroll
    for (int off = 8; off >= 1; off >>= 1) {
      float ov = __shfl_xor(rv[v], off);
      int   oi = __shfl_xor(ri[v], off);
      if (ov < rv[v] || (ov == rv[v] && oi < ri[v])) { rv[v] = ov; ri[v] = oi; }
    }
  }
  if (li == 0) {
#pragma unroll
    for (int v = 0; v < 8; ++v) {
      int r = mw * 16 + (g << 3) + v;    // C/D layout: M = g*8 + v
      redv[nw * MT + r] = rv[v];
      redi[nw * MT + r] = ri[v];
    }
  }
  __syncthreads();
  if (tid < MT) {
    float v0 = redv[tid], v1 = redv[MT + tid];
    int   i0 = redi[tid], i1 = redi[MT + tid];
    int kidx = (v1 < v0 || (v1 == v0 && i1 < i0)) ? i1 : i0;
    chosen[tid] = kidx;
    ((int*)out)[OUT_IDX + row0 + tid] = kidx;
    atomicAdd(&counts[kidx], 1.0f);
  }
  __syncthreads();

  // fp32 gather -> latents (== quantized), loss partials, dw scatter
  float lsum = 0.0f;
  for (int i = 0; i < (MT * Dn) / 256; ++i) {
    int idx = tid + i * 256;
    int r = idx >> 10, d = idx & (Dn - 1);
    int kidx = chosen[r];
    float qv = query[(long long)(row0 + r) * Dn + d];
    float ev = vq[(long long)d * Kn + kidx];
    out[OUT_LAT + (long long)(row0 + r) * Dn + d] = ev;
    float df = ev - qv;
    lsum = fmaf(df, df, lsum);
    atomicAdd(&dw[(long long)d * Kn + kidx], qv);   // distinct d per lane
  }
#pragma unroll
  for (int off = 16; off >= 1; off >>= 1) lsum += __shfl_xor(lsum, off);
  if (lane == 0) lred[w] = lsum;
  __syncthreads();
  if (tid == 0) {
    float t = 0.0f;
    for (int i = 0; i < 8; ++i) t += lred[i];
    atomicAdd(&out[OUT_LOSS], t * (CCOST_F / (float)Dn));
  }
}

// ---------------- kernel 5: new cluster sizes + global sum ----------------
__global__ void vq_cs(const float* __restrict__ cs, const float* __restrict__ counts,
                      float* __restrict__ out, float* __restrict__ nsum) {
  int k = blockIdx.x * 256 + threadIdx.x;
  float v = cs[k] * DECAY_F + counts[k] * (1.0f - DECAY_F);
  out[OUT_CS + k] = v;
  float s = v;
#pragma unroll
  for (int off = 16; off >= 1; off >>= 1) s += __shfl_xor(s, off);
  __shared__ float sh[8];
  int w = threadIdx.x >> 5, lane = threadIdx.x & 31;
  if (lane == 0) sh[w] = s;
  __syncthreads();
  if (threadIdx.x == 0) {
    float t = 0.0f;
    for (int i = 0; i < 8; ++i) t += sh[i];
    atomicAdd(nsum, t);
  }
}

// ---------------- kernel 6: EMA finalize (memory-bound) ----------------
__global__ void vq_fin(const float* __restrict__ ema, const float* __restrict__ dw,
                       const float* __restrict__ nsum, float* __restrict__ out) {
  long long i = (long long)blockIdx.x * 256 + threadIdx.x;
  if (i >= DK) return;
  int k = (int)(i & (Kn - 1));
  float ne = ema[i] * DECAY_F + dw[i] * (1.0f - DECAY_F);
  out[OUT_EMA + i] = ne;
  float n   = *nsum;
  float ncs = out[OUT_CS + k];
  float up  = (ncs + EPS_F) / (n + (float)Kn * EPS_F) * n;
  out[OUT_VQ + i] = ne / up;
}

// ---------------- launcher ----------------
extern "C" void kernel_launch(void* const* d_in, const int* in_sizes, int n_in,
                              void* d_out, int out_size, void* d_ws, size_t ws_size,
                              hipStream_t stream) {
  const float* query = (const float*)d_in[0];
  const float* vq    = (const float*)d_in[1];
  const float* ema   = (const float*)d_in[2];
  const float* cs    = (const float*)d_in[3];
  float* out = (float*)d_out;
  char*  wsb = (char*)d_ws;                       // needs ~84 MB
  float*    dw     = (float*)(wsb + WSB_DW);
  float*    counts = (float*)(wsb + WSB_CNT);
  float*    nsum   = (float*)(wsb + WSB_NS);
  float*    enorm  = (float*)(wsb + WSB_EN);
  uint16_t* bfE    = (uint16_t*)(wsb + WSB_BFE);
  uint16_t* bfQ    = (uint16_t*)(wsb + WSB_BFQ);

  long long zt = DK + Kn + 1;
  vq_zero <<<(int)((zt + 255) / 256), 256, 0, stream>>>((float*)wsb, out);
  vq_enorm<<<Kn / 256, 256, 0, stream>>>(vq, enorm);
  vq_cvtq <<<(int)(((long long)Nn * Dn / 2 + 255) / 256), 256, 0, stream>>>(query, bfQ);
  vq_cvte <<<(Kn / 64) * (Dn / 64), 256, 0, stream>>>(vq, bfE);

  size_t shmem = (size_t)MT * QSTR * 2
               + 2 * MT * sizeof(float) + 2 * MT * sizeof(int)
               + MT * sizeof(int) + 8 * sizeof(float);
  vq_main <<<Nn / MT, 256, shmem, stream>>>(query, vq, bfQ, bfE, enorm, out, dw, counts);

  vq_cs  <<<Kn / 256, 256, 0, stream>>>(cs, counts, out, nsum);
  vq_fin <<<(int)((DK + 255) / 256), 256, 0, stream>>>(ema, dw, nsum, out);
}
